// SelfAttentionBlock_18013092839929
// MI455X (gfx1250) — matrix operand order repaired and checked by reading, not compiled, not run on hardware
//
#include <hip/hip_runtime.h>

#define Bn 256
#define Cn 512
#define Sn 256
#define NH 8
#define DH 64
#define NG 32

typedef __attribute__((ext_vector_type(16))) __bf16 v16bf;
typedef __attribute__((ext_vector_type(8)))  __bf16 v8bf;
typedef __attribute__((ext_vector_type(4)))  __bf16 v4bf;
typedef __attribute__((ext_vector_type(8)))  float  v8f;

__device__ inline v16bf load_frag16(const __bf16* p0, const __bf16* p1) {
  v8bf a = *(const v8bf*)p0;
  v8bf b = *(const v8bf*)p1;
  return __builtin_shufflevector(a, b, 0,1,2,3,4,5,6,7,8,9,10,11,12,13,14,15);
}

__device__ inline v8f wmma_bf16(v16bf a, v16bf b, v8f c) {
  // D = A(16x32 bf16) * B(32x16 bf16) + C(16x16 f32)
  return __builtin_amdgcn_wmma_f32_16x16x32_bf16(false, a, false, b, (short)0, c, false, false);
}

// ---------------------------------------------------------------------------
// Kernel 0a: convert fp32 weights (K x N row-major) to transposed bf16 WT[N][K]
// ---------------------------------------------------------------------------
__global__ void k_wconv(const float* __restrict__ Wq, const float* __restrict__ Wk,
                        const float* __restrict__ Wv, const float* __restrict__ Wf,
                        __bf16* __restrict__ WT) {
  int idx = blockIdx.x * 256 + threadIdx.x;
  int m   = idx >> 18;
  int r   = idx & (Cn * Cn - 1);
  int eo  = r >> 9;
  int ei  = r & 511;
  const float* W = (m == 0) ? Wq : (m == 1) ? Wk : (m == 2) ? Wv : Wf;
  WT[(size_t)m * Cn * Cn + r] = (__bf16)W[ei * Cn + eo];
}

// ---------------------------------------------------------------------------
// Kernel 0b: LDS-tiled transpose x [B,C,S] fp32 -> xT [B,S,C] bf16
// ---------------------------------------------------------------------------
__global__ void k_xconv(const float* __restrict__ x, __bf16* __restrict__ xT) {
  __shared__ float tile[32][33];
  const int b   = blockIdx.z;
  const int ct0 = blockIdx.y * 32;
  const int st0 = blockIdx.x * 32;
  const int tx  = threadIdx.x & 31;
  const int ty  = threadIdx.x >> 5;             // 0..7
  for (int i = 0; i < 32; i += 8)
    tile[ty + i][tx] = x[((size_t)b * Cn + ct0 + ty + i) * Sn + st0 + tx];
  __syncthreads();
  for (int i = 0; i < 32; i += 8)
    xT[((size_t)b * Sn + st0 + ty + i) * Cn + ct0 + tx] = (__bf16)tile[tx][ty + i];
}

// ---------------------------------------------------------------------------
// Kernel 1: fused QKV projection + attention, one workgroup per (head, batch).
// Dynamic LDS layout (bytes):
//   q_s    [256][64] bf16      @ 0       (32768)
//   k_s    [256][64] bf16      @ 32768   (32768)
//   vT_s   [64][256] bf16      @ 65536   (32768)
//   phase A: xs [64][512] bf16 @ 98304   (65536)  <- async-to-LDS staging
//   phase B: logits 8w x [16][256] f32 @ 98304 (131072)  (aliases xs)
//   probs  8 waves x [16][256] bf16    @ 229376 (65536)
// total = 294912 bytes (< 320KB CDNA5 WGP LDS)
// ---------------------------------------------------------------------------
#define XS_OFF 98304u

__global__ void k_attn(const __bf16* __restrict__ xT,
                       const float* __restrict__ bq, const float* __restrict__ bk,
                       const float* __restrict__ bv,
                       const __bf16* __restrict__ WTq, const __bf16* __restrict__ WTk,
                       const __bf16* __restrict__ WTv,
                       __bf16* __restrict__ o_buf) {
  extern __shared__ char smem[];
  __bf16* q_s  = (__bf16*)(smem);
  __bf16* k_s  = (__bf16*)(smem + 32768);
  __bf16* vT_s = (__bf16*)(smem + 65536);
  __bf16* xs   = (__bf16*)(smem + XS_OFF);   // phase A only
  float*  lgA  = (float*) (smem + XS_OFF);   // phase B only (aliases xs)
  __bf16* pbA  = (__bf16*)(smem + 229376);

  const int head = blockIdx.x;
  const int b    = blockIdx.y;
  const int tid  = threadIdx.x;
  const int wave = tid >> 5;
  const int lane = tid & 31;
  const int row  = lane & 15;       // A-row / B-col / D-col lane mapping
  const int krow = lane >> 4;       // fragment K sub-block select

  // ---------------- Phase A: QKV projection, 64 S-rows per sweep ----------
  for (int sc = 0; sc < Sn / 64; ++sc) {
    const int s0 = sc * 64;
    // stage xs [64][512] bf16 = 64KB via async global->LDS DMA (16B/lane/op)
    {
      const char* src = (const char*)(xT + ((size_t)b * Sn + s0) * Cn);
      for (int t = tid; t < 4096; t += 256) {
        unsigned    ldso = XS_OFF + (unsigned)t * 16u;
        const char* g    = src + (size_t)t * 16u;
        asm volatile("global_load_async_to_lds_b128 %0, %1, off"
                     :: "v"(ldso), "v"(g) : "memory");
      }
      asm volatile("s_wait_asynccnt 0" ::: "memory");
    }
    __syncthreads();

    // 12 output tiles: waves 0..3 -> q+v (nt=wave), waves 4..7 -> k (nt=wave-4).
    // Scalarized branch (readfirstlane) keeps EXEC all-1s around every WMMA.
    const int nt0 = wave & 3;
    const int ng  = head * DH + nt0 * 16 + row;   // global output column
    const int d   = nt0 * 16 + row;               // column within head
    if (__builtin_amdgcn_readfirstlane(wave) < 4) {
      v8f accq[4] = {}, accv[4] = {};
      for (int kt = 0; kt < Cn / 32; ++kt) {
        const __bf16* bpq = WTq + (size_t)ng * Cn + kt * 32 + krow * 8;
        const __bf16* bpv = WTv + (size_t)ng * Cn + kt * 32 + krow * 8;
        v16bf b0 = load_frag16(bpq, bpq + 16);
        v16bf b1 = load_frag16(bpv, bpv + 16);
        v16bf a[4];
        for (int rt = 0; rt < 4; ++rt) {
          const __bf16* ap = xs + (rt * 16 + row) * Cn + kt * 32 + krow * 8;
          a[rt] = load_frag16(ap, ap + 16);
        }
        for (int rt = 0; rt < 4; ++rt) accq[rt] = wmma_bf16(a[rt], b0, accq[rt]);
        for (int rt = 0; rt < 4; ++rt) accv[rt] = wmma_bf16(a[rt], b1, accv[rt]);
      }
      const float biaq = bq[ng], biav = bv[ng];
      for (int rt = 0; rt < 4; ++rt)
        for (int i = 0; i < 8; ++i) {
          const int sl = s0 + rt * 16 + i + krow * 8;
          q_s [sl * DH + d] = (__bf16)(accq[rt][i] + biaq);
          vT_s[d * Sn + sl] = (__bf16)(accv[rt][i] + biav);
        }
    } else {
      v8f acck[4] = {};
      for (int kt = 0; kt < Cn / 32; ++kt) {
        const __bf16* bpk = WTk + (size_t)ng * Cn + kt * 32 + krow * 8;
        v16bf b0 = load_frag16(bpk, bpk + 16);
        v16bf a[4];
        for (int rt = 0; rt < 4; ++rt) {
          const __bf16* ap = xs + (rt * 16 + row) * Cn + kt * 32 + krow * 8;
          a[rt] = load_frag16(ap, ap + 16);
        }
        for (int rt = 0; rt < 4; ++rt) acck[rt] = wmma_bf16(a[rt], b0, acck[rt]);
      }
      const float biak = bk[ng];
      for (int rt = 0; rt < 4; ++rt)
        for (int i = 0; i < 8; ++i) {
          const int sl = s0 + rt * 16 + i + krow * 8;
          k_s[sl * DH + d] = (__bf16)(acck[rt][i] + biak);
        }
    }
    __syncthreads();
  }

  // ---------------- Phase B: attention (2 q-tiles per wave) ----------------
  float*  lg = lgA + wave * 16 * Sn;
  __bf16* pb = pbA + wave * 16 * Sn;
  const float scale = 0.125f;  // 1/sqrt(64)

  for (int qt = wave; qt < Sn / 16; qt += 8) {
    const int s0 = qt * 16;
    // q fragments (K = 64 -> 2 fragments), kept in registers
    const __bf16* qp0 = q_s + (s0 + row) * DH + krow * 8;
    const __bf16* qp1 = q_s + (s0 + row) * DH + 32 + krow * 8;
    v16bf aq0 = load_frag16(qp0, qp0 + 16);
    v16bf aq1 = load_frag16(qp1, qp1 + 16);

    // logits = scale * q @ k^T
    for (int nt = 0; nt < Sn / 16; ++nt) {
      const int t = nt * 16 + row;
      const __bf16* bp0 = k_s + t * DH + krow * 8;
      const __bf16* bp1 = k_s + t * DH + 32 + krow * 8;
      v16bf b0 = load_frag16(bp0, bp0 + 16);
      v16bf b1 = load_frag16(bp1, bp1 + 16);
      v8f acc = {};
      acc = wmma_bf16(aq0, b0, acc);
      acc = wmma_bf16(aq1, b1, acc);
      for (int i = 0; i < 8; ++i)
        lg[(i + krow * 8) * Sn + nt * 16 + row] = acc[i] * scale;
    }

    // softmax: all 32 lanes active; lane&15 = row, lane>>4 = 128-col half.
    // Half-row partials combined with a cross-lane XOR-16 shuffle.
    {
      float* lr = lg + row * Sn;
      float4* lr4 = (float4*)(lr + krow * 128);        // 32 x float4
      float m = -1e30f;
      for (int c = 0; c < 32; ++c) {
        float4 v = lr4[c];
        m = fmaxf(m, fmaxf(fmaxf(v.x, v.y), fmaxf(v.z, v.w)));
      }
      m = fmaxf(m, __shfl_xor(m, 16, 32));
      float ssum = 0.f;
      for (int c = 0; c < 32; ++c) {
        float4 v = lr4[c];
        v.x = __expf(v.x - m); v.y = __expf(v.y - m);
        v.z = __expf(v.z - m); v.w = __expf(v.w - m);
        lr4[c] = v;
        ssum += (v.x + v.y) + (v.z + v.w);
      }
      ssum += __shfl_xor(ssum, 16, 32);
      const float inv = 1.f / ssum;
      __bf16* pr = pb + row * Sn + krow * 128;
      for (int c = 0; c < 32; ++c) {
        float4 v = lr4[c];
        v4bf o;
        o[0] = (__bf16)(v.x * inv); o[1] = (__bf16)(v.y * inv);
        o[2] = (__bf16)(v.z * inv); o[3] = (__bf16)(v.w * inv);
        *(v4bf*)(pr + c * 4) = o;
      }
    }

    // o = probs @ v : K = 256 -> 8 fragments; A shared across all 4 d-tiles
    v8f acc[4] = {};
    for (int kt = 0; kt < Sn / 32; ++kt) {
      const __bf16* ap = pb + row * Sn + kt * 32 + krow * 8;
      v16bf a = load_frag16(ap, ap + 16);
      v16bf bb[4];
      for (int ntd = 0; ntd < DH / 16; ++ntd) {
        const __bf16* bp = vT_s + (ntd * 16 + row) * Sn + kt * 32 + krow * 8;
        bb[ntd] = load_frag16(bp, bp + 16);
      }
      for (int ntd = 0; ntd < DH / 16; ++ntd)
        acc[ntd] = wmma_bf16(a, bb[ntd], acc[ntd]);
    }
    for (int ntd = 0; ntd < DH / 16; ++ntd) {
      const int d = ntd * 16 + row;
      for (int i = 0; i < 8; ++i) {
        const int sl = i + krow * 8;
        o_buf[((size_t)b * Sn + s0 + sl) * Cn + head * DH + d] = (__bf16)acc[ntd][i];
      }
    }
  }
}

// ---------------------------------------------------------------------------
// Kernel 2: y = o @ Wf + bf + x (residual), fp32 into d_out [B,C,S].
// 32-row M-blocks: each B fragment feeds two row-tiles (8 accumulators/wave).
// ---------------------------------------------------------------------------
__global__ void k_proj(const __bf16* __restrict__ o_buf, const __bf16* __restrict__ WTf,
                       const float* __restrict__ bf, const float* __restrict__ x,
                       float* __restrict__ out) {
  const int b  = blockIdx.x >> 3;
  const int s0 = (blockIdx.x & 7) * 32;
  const int tid  = threadIdx.x;
  const int wave = tid >> 5;
  const int lane = tid & 31;
  const int row  = lane & 15;
  const int krow = lane >> 4;
  const int ntb  = wave * 4;

  const __bf16* arow0 = o_buf + ((size_t)b * Sn + s0 + row) * Cn;
  const __bf16* arow1 = arow0 + (size_t)16 * Cn;
  v8f acc[2][4] = {};
  for (int kt = 0; kt < Cn / 32; ++kt) {
    const __bf16* ap0 = arow0 + kt * 32 + krow * 8;
    const __bf16* ap1 = arow1 + kt * 32 + krow * 8;
    v16bf a0 = load_frag16(ap0, ap0 + 16);
    v16bf a1 = load_frag16(ap1, ap1 + 16);
    v16bf bb[4];
    for (int j = 0; j < 4; ++j) {
      const __bf16* bp = WTf + (size_t)((ntb + j) * 16 + row) * Cn + kt * 32 + krow * 8;
      bb[j] = load_frag16(bp, bp + 16);
    }
    for (int j = 0; j < 4; ++j) acc[0][j] = wmma_bf16(a0, bb[j], acc[0][j]);
    for (int j = 0; j < 4; ++j) acc[1][j] = wmma_bf16(a1, bb[j], acc[1][j]);
  }
  for (int j = 0; j < 4; ++j) {
    const int n = (ntb + j) * 16 + row;
    const float bia = bf[n];
    for (int rt = 0; rt < 2; ++rt)
      for (int i = 0; i < 8; ++i) {
        const int sl = rt * 16 + i + krow * 8;
        const size_t xi = ((size_t)b * Cn + n) * Sn + s0 + sl;
        out[xi] = acc[rt][j][i] + bia + x[xi];
      }
  }
}

// ---------------------------------------------------------------------------
// Kernel 3: in-place GroupNorm over d_out [B,C,S]; one block per (b, group)
// ---------------------------------------------------------------------------
__global__ void k_gnorm(float* __restrict__ out, const float* __restrict__ gw,
                        const float* __restrict__ gb) {
  const int b  = blockIdx.x >> 5;
  const int g  = blockIdx.x & 31;
  const int c0 = g * (Cn / NG);
  const int tid = threadIdx.x;
  __shared__ float red[256], red2[256];

  float s = 0.f, s2 = 0.f;
  for (int k = 0; k < 16; ++k) {
    const int idx = k * 256 + tid;
    const int c = idx >> 8, sp = idx & 255;
    const float v = out[((size_t)b * Cn + c0 + c) * Sn + sp];
    s += v; s2 += v * v;
  }
  red[tid] = s; red2[tid] = s2;
  __syncthreads();
  for (int off = 128; off > 0; off >>= 1) {
    if (tid < off) { red[tid] += red[tid + off]; red2[tid] += red2[tid + off]; }
    __syncthreads();
  }
  const float mu  = red[0] * (1.f / 4096.f);
  const float var = red2[0] * (1.f / 4096.f) - mu * mu;
  const float rs  = rsqrtf(var + 1e-5f);
  for (int k = 0; k < 16; ++k) {
    const int idx = k * 256 + tid;
    const int c = idx >> 8, sp = idx & 255;
    const size_t a = ((size_t)b * Cn + c0 + c) * Sn + sp;
    out[a] = (out[a] - mu) * rs * gw[c0 + c] + gb[c0 + c];
  }
}

// ---------------------------------------------------------------------------
extern "C" void kernel_launch(void* const* d_in, const int* in_sizes, int n_in,
                              void* d_out, int out_size, void* d_ws, size_t ws_size,
                              hipStream_t stream) {
  const float* x    = (const float*)d_in[0];
  const float* Wq   = (const float*)d_in[1];
  const float* bq   = (const float*)d_in[2];
  const float* Wk   = (const float*)d_in[3];
  const float* bk   = (const float*)d_in[4];
  const float* Wv   = (const float*)d_in[5];
  const float* bv   = (const float*)d_in[6];
  const float* Wf   = (const float*)d_in[7];
  const float* bf   = (const float*)d_in[8];
  const float* gn_w = (const float*)d_in[9];
  const float* gn_b = (const float*)d_in[10];
  float* out = (float*)d_out;

  __bf16* WT    = (__bf16*)d_ws;                                       // 4x512x512 bf16
  __bf16* xT    = (__bf16*)((char*)d_ws + (size_t)4 * Cn * Cn * 2);    // B x S x C bf16
  __bf16* o_buf = (__bf16*)((char*)xT + (size_t)Bn * Sn * Cn * 2);     // B x S x C bf16

  // 0a) weight convert + transpose -> bf16
  k_wconv<<<(4 * Cn * Cn) / 256, 256, 0, stream>>>(Wq, Wk, Wv, Wf, WT);

  // 0b) x transpose + convert -> bf16 [B,S,C]
  k_xconv<<<dim3(Sn / 32, Cn / 32, Bn), 256, 0, stream>>>(x, xT);

  // 1) fused QKV + attention per (head, batch), async-to-LDS staging
  k_attn<<<dim3(NH, Bn), 256, 294912, stream>>>(
      xT, bq, bk, bv, WT, WT + Cn * Cn, WT + 2 * Cn * Cn, o_buf);

  // 2) output projection + residual -> d_out (fp32, [B,C,S])
  k_proj<<<Bn * (Sn / 32), 256, 0, stream>>>(o_buf, WT + 3 * Cn * Cn, bf, x, out);

  // 3) in-place GroupNorm
  k_gnorm<<<Bn * NG, 256, 0, stream>>>(out, gn_w, gn_b);
}
